// MultiHeadAttention_44942537785683
// MI455X (gfx1250) — matrix operand order repaired and checked by reading, not compiled, not run on hardware
//
#include <hip/hip_runtime.h>
#include <stdint.h>

// ---------------------------------------------------------------------------
// MHA forward for MI455X (gfx1250).  All matmuls: v_wmma_f32_16x16x32_bf16.
// B=2, S=2048, E=1024, H=16, D=64.  SCALE = 1/sqrt(64) = 0.125
//   1) cast f32 -> bf16 (query/key/value, Wq/Wk/Wv/Wo)
//   2) GEMM+bias (LDS double-buffered; X tile via async-to-LDS, W tile via TDM)
//      wave tile 32x64: 8 WMMA per 32-k step sharing one set of B fragments
//   3) flash attention (K/V tiles staged per block via TDM / async-to-LDS)
//   4) GEMM+bias -> f32 d_out
// Workspace: 3*8MB X + 4*2MB W + 8MB Q + 8MB K + 8MB Vt + 8MB attn = 64 MB.
// ---------------------------------------------------------------------------

typedef __attribute__((ext_vector_type(4)))  unsigned int u32x4;
typedef __attribute__((ext_vector_type(4)))  int          i32x4;
typedef __attribute__((ext_vector_type(8)))  int          i32x8;
typedef __attribute__((ext_vector_type(16))) __bf16       bf16x16;
typedef __attribute__((ext_vector_type(8)))  float        f32x8;

union FragBF {           // 16 bf16 = 32B = two 16-byte loads
    u32x4   u[2];
    bf16x16 v;
};

#define NUM_HEADS 16
#define HEAD_DIM  64
#define EMBED     1024
#define SEQ       2048
#define MROWS     4096          // B*S
#define ATT_SCALE 0.125f

// ---- gfx1250 feature probes ------------------------------------------------
#if __has_builtin(__builtin_amdgcn_global_load_async_to_lds_b128)
#define USE_ASYNC_LDS 1
#else
#define USE_ASYNC_LDS 0
#endif

#if __has_builtin(__builtin_amdgcn_tensor_load_to_lds)
#define USE_TDM 1
#else
#define USE_TDM 0
#endif

#if USE_TDM && __has_include(<hip/amd_detail/amd_gfx1250_TDM.h>)
#define TDM_SIX_ARGS 1
#else
#define TDM_SIX_ARGS 0
#endif

// async-to-LDS builtin parameter types (from hipcc diagnostic):
//   param0: non-const GCC-vector int4 pointer in AS(1), param1: AS(3)
typedef int v4i_gcc __attribute__((vector_size(16)));
typedef __attribute__((address_space(1))) v4i_gcc* gv4i_p;
typedef __attribute__((address_space(3))) v4i_gcc* lv4i_p;

// 16B global -> LDS, async when available (ASYNCcnt), else register bounce
__device__ __forceinline__ void copy16_g2l(const unsigned short* g, unsigned short* l) {
#if USE_ASYNC_LDS
    __builtin_amdgcn_global_load_async_to_lds_b128(
        (gv4i_p)(uintptr_t)g, (lv4i_p)(unsigned int)(uintptr_t)l, 0, 0);
#else
    *(u32x4*)l = *(const u32x4*)g;
#endif
}

__device__ __forceinline__ void wait_async_copies() {
#if USE_ASYNC_LDS
#if __has_builtin(__builtin_amdgcn_s_wait_asynccnt)
    __builtin_amdgcn_s_wait_asynccnt(0);
#else
    asm volatile("s_wait_asynccnt 0" ::: "memory");
#endif
#endif
}

#if USE_TDM
// 2-D TDM tile load (ISA 8.3/8.4): tile_d0 contiguous elements per row,
// tile_d1 rows, row stride stride0 (all in bf16 elements).  Groups 2/3 = 0
// (<=2-D tensor).  Issue from ONE wave (EXEC ignored; one instr per wave).
__device__ __forceinline__ void tdm_load_2d(unsigned int lds_off,
                                            const unsigned short* gsrc,
                                            int tile_d0, int tile_d1,
                                            int t_dim0, int t_dim1, int stride0) {
    const unsigned long long ga = (unsigned long long)(uintptr_t)gsrc;
    u32x4 g0;
    g0[0] = 1u;                                              // count=1 (valid user D#)
    g0[1] = lds_off;                                         // lds_addr (bytes)
    g0[2] = (unsigned int)(ga & 0xFFFFFFFFull);              // global_addr[31:0]
    g0[3] = (unsigned int)((ga >> 32) & 0x1FFFFFFull)        // global_addr[56:32]
          | (2u << 30);                                      // type = 2 ("image")
    i32x8 g1;
    g1[0] = 0x00010000;                                      // data_size=1 -> 2 bytes
    g1[1] = (int)((unsigned)t_dim0 << 16);                   // tensor_dim0 lo16
    g1[2] = (int)(((unsigned)t_dim0 >> 16) | ((unsigned)t_dim1 << 16));
    g1[3] = (int)(((unsigned)t_dim1 >> 16) | ((unsigned)tile_d0 << 16));
    g1[4] = tile_d1 & 0xFFFF;                                // tile_dim1 (tile_dim2=0)
    g1[5] = stride0;                                         // tensor_dim0_stride lo32
    g1[6] = 0;
    g1[7] = 0;
    i32x4 z4 = (i32x4)0;
#if TDM_SIX_ARGS
    i32x8 z8 = (i32x8)0;
    __builtin_amdgcn_tensor_load_to_lds(g0, g1, z4, z4, z8, 0);
#else
    __builtin_amdgcn_tensor_load_to_lds(g0, g1, z4, z4, 0);
#endif
}
__device__ __forceinline__ void wait_tdm() {
    __builtin_amdgcn_s_wait_tensorcnt(0);
}
#endif

// round-to-nearest-even f32 -> bf16 bits
__device__ __forceinline__ unsigned short f32_to_bf16_bits(float f) {
    unsigned int u = __builtin_bit_cast(unsigned int, f);
    u += 0x7FFFu + ((u >> 16) & 1u);
    return (unsigned short)(u >> 16);
}

// ---------------------------------------------------------------------------
// 1) cast kernel: 8 f32 -> 8 bf16 per thread
// ---------------------------------------------------------------------------
__global__ void mha_cast_bf16(const float* __restrict__ src,
                              unsigned short* __restrict__ dst, int n8) {
    int i = blockIdx.x * blockDim.x + threadIdx.x;
    if (i >= n8) return;
    union { unsigned short s[8]; u32x4 v; } pk;
    const float* p = src + 8ull * i;
#pragma unroll
    for (int j = 0; j < 8; ++j) pk.s[j] = f32_to_bf16_bits(p[j]);
    *(u32x4*)(dst + 8ull * i) = pk.v;
}

// ---------------------------------------------------------------------------
// 2/4) GEMM: C[4096,1024] = X @ W^T (+bias).  W is [N,K] row-major.
//   Block 256 thr (8 waves); block tile 128 rows x 128 cols.
//   Wave tile 32x64: 2 A fragments share 4 B fragments -> 8 WMMA / k-step.
//   K-chunks of 32, double-buffered in LDS:
//     X tile 128x32: per-lane async-to-LDS b128   (ASYNCcnt)
//     W tile 128x32: one TDM descriptor, wave 0   (TENSORcnt)
//   mode 0: f32 out[row*1024+col]                 (O projection)
//   mode 1: bf16 out[((b*16+h)*2048+s)*64+d]      (Q, K head-major)
//   mode 2: bf16 out[((b*16+h)*64+d)*2048+s]      (V transposed)
// ---------------------------------------------------------------------------
__global__ void mha_gemm_bf16_wmma(const unsigned short* __restrict__ X,
                                   const unsigned short* __restrict__ W,
                                   const float* __restrict__ bias,
                                   void* __restrict__ out, int mode) {
    __shared__ unsigned short Xs[2][128][32];    // 2 x 8 KB
    __shared__ unsigned short Ws[2][128][32];    // 2 x 8 KB

    const int tid  = threadIdx.x;
    const int lane = tid & 31;
    const int wave = tid >> 5;
    const int half = lane >> 4;                  // ISA 7.12.2 lane halves
    const int l16  = lane & 15;
    const int rowBase0 = blockIdx.x * 128;
    const int colBase0 = blockIdx.y * 128;
    const int rowBase  = rowBase0 + (wave & 3) * 32;
    const int colBase  = colBase0 + (wave >> 2) * 64;

    auto stage = [&](int buf, int k0) {
        {   // X tile: 128 rows x 4 chunks of 8 bf16 = 512 x 16B -> 2 per thread
#pragma unroll
            for (int i = 0; i < 2; ++i) {
                const int c = tid + i * 256;
                const int r = c >> 2, ch = c & 3;
                copy16_g2l(X + (size_t)(rowBase0 + r) * EMBED + k0 + ch * 8,
                           &Xs[buf][r][ch * 8]);
            }
        }
#if USE_TDM
        if (wave == 0)
            tdm_load_2d((unsigned int)(uintptr_t)&Ws[buf][0][0],
                        W + (size_t)colBase0 * EMBED + k0,
                        /*tile*/ 32, 128, /*tensor*/ EMBED, EMBED, /*stride*/ EMBED);
#else
        {   // W tile: 128 rows x 4 chunks, 512 x 16B -> 2 per thread
#pragma unroll
            for (int i = 0; i < 2; ++i) {
                const int c = tid + i * 256;
                const int r = c >> 2, ch = c & 3;
                copy16_g2l(W + (size_t)(colBase0 + r) * EMBED + k0 + ch * 8,
                           &Ws[buf][r][ch * 8]);
            }
        }
#endif
    };
    auto stage_wait = [&]() {
        wait_async_copies();
#if USE_TDM
        if (wave == 0) wait_tdm();
#endif
        __syncthreads();
    };

    f32x8 acc[8] = {};                           // [i*4+j]: i=row tile, j=col tile
    int buf = 0;
    stage(0, 0);
    stage_wait();

    for (int k0 = 0; k0 < EMBED; k0 += 32) {
        if (k0 + 32 < EMBED) stage(buf ^ 1, k0 + 32);

        FragBF a[2];                             // A 16x32: lane=row, k split 8/8
#pragma unroll
        for (int i = 0; i < 2; ++i) {
            const unsigned short* ap =
                &Xs[buf][(wave & 3) * 32 + i * 16 + l16][half * 8];
            a[i].u[0] = *(const u32x4*)ap;
            a[i].u[1] = *(const u32x4*)(ap + 16);
        }
#pragma unroll
        for (int j = 0; j < 4; ++j) {            // B 32x16: lane=col, k split 8/8
            FragBF b;
            const unsigned short* bp =
                &Ws[buf][(wave >> 2) * 64 + j * 16 + l16][half * 8];
            b.u[0] = *(const u32x4*)bp;
            b.u[1] = *(const u32x4*)(bp + 16);
#pragma unroll
            for (int i = 0; i < 2; ++i)
                acc[i * 4 + j] = __builtin_amdgcn_wmma_f32_16x16x32_bf16(
                    false, a[i].v, false, b.v, (short)0, acc[i * 4 + j], false, false);
        }
        stage_wait();
        buf ^= 1;
    }

#pragma unroll
    for (int i = 0; i < 2; ++i)
#pragma unroll
        for (int j = 0; j < 4; ++j) {
            const int col = colBase + j * 16 + l16;
            const float bb = bias[col];
#pragma unroll
            for (int r = 0; r < 8; ++r) {        // C layout: lane holds rows half*8+r
                const int row = rowBase + i * 16 + half * 8 + r;
                const float v = acc[i * 4 + j][r] + bb;
                if (mode == 0) {
                    ((float*)out)[(size_t)row * EMBED + col] = v;
                } else {
                    const int b = row >> 11, s = row & 2047;
                    const int h = col >> 6,  d = col & 63;
                    const unsigned short hv = f32_to_bf16_bits(v);
                    if (mode == 1)
                        ((unsigned short*)out)[((size_t)(b * NUM_HEADS + h) * SEQ + s) * HEAD_DIM + d] = hv;
                    else
                        ((unsigned short*)out)[((size_t)(b * NUM_HEADS + h) * HEAD_DIM + d) * SEQ + s] = hv;
                }
            }
        }
}

// ---------------------------------------------------------------------------
// 3) flash attention.  grid = (B*H = 32, S/128 = 16), 256 thr (8 waves).
//    Each wave: 16 query rows x D=64, online softmax over 2048 keys in
//    32-key steps.  K [bh,s,d], Vt [bh,d,s] bf16.  K/V tiles are staged
//    ONCE per block (shared by all 8 waves), double-buffered.
// ---------------------------------------------------------------------------
__global__ void mha_attention_wmma(const unsigned short* __restrict__ Q,
                                   const unsigned short* __restrict__ K,
                                   const unsigned short* __restrict__ Vt,
                                   unsigned short* __restrict__ Out) {
    __shared__ unsigned short Ks[2][32][64];     // 2 x 4 KB  (keys x d)
    __shared__ unsigned short Vs[2][64][32];     // 2 x 4 KB  (d x keys)
    __shared__ unsigned short Ps[8][16][32];     // per-wave P staging, 8 KB

    const int tid  = threadIdx.x;
    const int lane = tid & 31;
    const int wave = tid >> 5;
    const int half = lane >> 4;
    const int l16  = lane & 15;
    const int bh   = blockIdx.x;                 // b*16 + h
    const int qTile = blockIdx.y * 128 + wave * 16;

    const unsigned short* Qh = Q  + (size_t)bh * SEQ * HEAD_DIM;
    const unsigned short* Kh = K  + (size_t)bh * SEQ * HEAD_DIM;
    const unsigned short* Vh = Vt + (size_t)bh * HEAD_DIM * SEQ;

    auto stage = [&](int buf, int kb) {
#if USE_TDM
        if (wave == 0) {
            tdm_load_2d((unsigned int)(uintptr_t)&Ks[buf][0][0],
                        Kh + (size_t)kb * HEAD_DIM,
                        /*tile*/ HEAD_DIM, 32, /*tensor*/ HEAD_DIM, SEQ, HEAD_DIM);
            tdm_load_2d((unsigned int)(uintptr_t)&Vs[buf][0][0],
                        Vh + kb,
                        /*tile*/ 32, HEAD_DIM, /*tensor*/ SEQ, HEAD_DIM, SEQ);
        }
#else
        {   // K tile: 32 keys x 8 chunks = 256 x 16B
            const int r = tid >> 3, c = tid & 7;
            copy16_g2l(Kh + (size_t)(kb + r) * HEAD_DIM + c * 8, &Ks[buf][r][c * 8]);
        }
        {   // V tile: 64 d x 4 chunks = 256 x 16B
            const int r = tid >> 2, c = tid & 3;
            copy16_g2l(Vh + (size_t)r * SEQ + kb + c * 8, &Vs[buf][r][c * 8]);
        }
#endif
    };
    auto stage_wait = [&]() {
        wait_async_copies();
#if USE_TDM
        if (wave == 0) wait_tdm();
#endif
        __syncthreads();
    };

    // Q fragments (d=0..31, 32..63), register-resident for the whole loop
    FragBF a0, a1;
    {
        const unsigned short* qp = Qh + (size_t)(qTile + l16) * HEAD_DIM;
        a0.u[0] = *(const u32x4*)(qp + half * 8);
        a0.u[1] = *(const u32x4*)(qp + 16 + half * 8);
        a1.u[0] = *(const u32x4*)(qp + 32 + half * 8);
        a1.u[1] = *(const u32x4*)(qp + 48 + half * 8);
    }

    float m_i[8], l_i[8];
#pragma unroll
    for (int r = 0; r < 8; ++r) { m_i[r] = -1e30f; l_i[r] = 0.0f; }
    f32x8 acc[4] = {};

    int buf = 0;
    stage(0, 0);
    stage_wait();

    for (int kb = 0; kb < SEQ; kb += 32) {
        if (kb + 32 < SEQ) stage(buf ^ 1, kb + 32);

        // ---- scores: S[16x32] = Q @ K^T, B-fragments from LDS ----
        f32x8 s[2] = {};
#pragma unroll
        for (int t = 0; t < 2; ++t) {
            const unsigned short* kp = &Ks[buf][t * 16 + l16][0];
            FragBF b0, b1;
            b0.u[0] = *(const u32x4*)(kp + half * 8);
            b0.u[1] = *(const u32x4*)(kp + half * 8 + 16);
            b1.u[0] = *(const u32x4*)(kp + 32 + half * 8);
            b1.u[1] = *(const u32x4*)(kp + 32 + half * 8 + 16);
            s[t] = __builtin_amdgcn_wmma_f32_16x16x32_bf16(
                       false, a0.v, false, b0.v, (short)0, s[t], false, false);
            s[t] = __builtin_amdgcn_wmma_f32_16x16x32_bf16(
                       false, a1.v, false, b1.v, (short)0, s[t], false, false);
        }

        // ---- online softmax (rows in 16-lane halves; xor<16 stays inside) ----
        float alpha[8];
#pragma unroll
        for (int r = 0; r < 8; ++r) {
            float x0 = s[0][r] * ATT_SCALE;
            float x1 = s[1][r] * ATT_SCALE;
            float mx = fmaxf(x0, x1);
            mx = fmaxf(mx, __shfl_xor(mx, 1, 32));
            mx = fmaxf(mx, __shfl_xor(mx, 2, 32));
            mx = fmaxf(mx, __shfl_xor(mx, 4, 32));
            mx = fmaxf(mx, __shfl_xor(mx, 8, 32));
            const float mnew = fmaxf(m_i[r], mx);
            const float p0 = __expf(x0 - mnew);
            const float p1 = __expf(x1 - mnew);
            float rs = p0 + p1;
            rs += __shfl_xor(rs, 1, 32);
            rs += __shfl_xor(rs, 2, 32);
            rs += __shfl_xor(rs, 4, 32);
            rs += __shfl_xor(rs, 8, 32);
            alpha[r] = __expf(m_i[r] - mnew);
            l_i[r] = l_i[r] * alpha[r] + rs;
            m_i[r] = mnew;
            Ps[wave][half * 8 + r][l16]      = f32_to_bf16_bits(p0);
            Ps[wave][half * 8 + r][16 + l16] = f32_to_bf16_bits(p1);
        }
#pragma unroll
        for (int j = 0; j < 4; ++j)
#pragma unroll
            for (int r = 0; r < 8; ++r) acc[j][r] *= alpha[r];

        // wave-private P region: order cross-lane ds stores before reload
        asm volatile("s_wait_dscnt 0" ::: "memory");

        FragBF pa;                               // P as A-fragment 16x32
        pa.u[0] = *(const u32x4*)(&Ps[wave][l16][half * 8]);
        pa.u[1] = *(const u32x4*)(&Ps[wave][l16][16 + half * 8]);

        // ---- acc += P @ V, B-fragments from LDS ----
#pragma unroll
        for (int j = 0; j < 4; ++j) {
            const unsigned short* vp = &Vs[buf][j * 16 + l16][half * 8];
            FragBF bv;
            bv.u[0] = *(const u32x4*)vp;
            bv.u[1] = *(const u32x4*)(vp + 16);
            acc[j] = __builtin_amdgcn_wmma_f32_16x16x32_bf16(
                         false, pa.v, false, bv.v, (short)0, acc[j], false, false);
        }

        stage_wait();
        buf ^= 1;
    }

    // ---- normalize, store bf16 attn output [B,S,E] for the O projection ----
    float inv[8];
#pragma unroll
    for (int r = 0; r < 8; ++r) inv[r] = 1.0f / l_i[r];
    const int b = bh >> 4, h = bh & 15;
#pragma unroll
    for (int j = 0; j < 4; ++j)
#pragma unroll
        for (int r = 0; r < 8; ++r) {
            const int srow = qTile + half * 8 + r;
            const int d = j * 16 + l16;
            Out[((size_t)(b * SEQ + srow)) * EMBED + h * HEAD_DIM + d] =
                f32_to_bf16_bits(acc[j][r] * inv[r]);
        }
}

// ---------------------------------------------------------------------------
// launch
// ---------------------------------------------------------------------------
extern "C" void kernel_launch(void* const* d_in, const int* in_sizes, int n_in,
                              void* d_out, int out_size, void* d_ws, size_t ws_size,
                              hipStream_t stream) {
    (void)in_sizes; (void)n_in; (void)out_size; (void)ws_size;

    const float* q  = (const float*)d_in[0];
    const float* k  = (const float*)d_in[1];
    const float* v  = (const float*)d_in[2];
    const float* Wq = (const float*)d_in[3];
    const float* bq = (const float*)d_in[4];
    const float* Wk = (const float*)d_in[5];
    const float* bk = (const float*)d_in[6];
    const float* Wv = (const float*)d_in[7];
    const float* bv = (const float*)d_in[8];
    const float* Wo = (const float*)d_in[9];
    const float* bo = (const float*)d_in[10];

    char* ws = (char*)d_ws;
    const size_t SZ_X = (size_t)MROWS * EMBED * 2;   // 8 MB
    const size_t SZ_W = (size_t)EMBED * EMBED * 2;   // 2 MB
    unsigned short* Xq  = (unsigned short*)(ws);
    unsigned short* Xk  = (unsigned short*)(ws + SZ_X);
    unsigned short* Xv  = (unsigned short*)(ws + 2 * SZ_X);
    unsigned short* Wqb = (unsigned short*)(ws + 3 * SZ_X);
    unsigned short* Wkb = (unsigned short*)(ws + 3 * SZ_X + SZ_W);
    unsigned short* Wvb = (unsigned short*)(ws + 3 * SZ_X + 2 * SZ_W);
    unsigned short* Wob = (unsigned short*)(ws + 3 * SZ_X + 3 * SZ_W);
    unsigned short* Qh  = (unsigned short*)(ws + 3 * SZ_X + 4 * SZ_W);
    unsigned short* Kh  = (unsigned short*)(ws + 4 * SZ_X + 4 * SZ_W);
    unsigned short* Vt  = (unsigned short*)(ws + 5 * SZ_X + 4 * SZ_W);
    unsigned short* AO  = (unsigned short*)(ws + 6 * SZ_X + 4 * SZ_W);

    const int actN8 = MROWS * EMBED / 8;     // 524288
    const int wgtN8 = EMBED * EMBED / 8;     // 131072
    mha_cast_bf16<<<actN8 / 256, 256, 0, stream>>>(q,  Xq,  actN8);
    mha_cast_bf16<<<actN8 / 256, 256, 0, stream>>>(k,  Xk,  actN8);
    mha_cast_bf16<<<actN8 / 256, 256, 0, stream>>>(v,  Xv,  actN8);
    mha_cast_bf16<<<wgtN8 / 256, 256, 0, stream>>>(Wq, Wqb, wgtN8);
    mha_cast_bf16<<<wgtN8 / 256, 256, 0, stream>>>(Wk, Wkb, wgtN8);
    mha_cast_bf16<<<wgtN8 / 256, 256, 0, stream>>>(Wv, Wvb, wgtN8);
    mha_cast_bf16<<<wgtN8 / 256, 256, 0, stream>>>(Wo, Wob, wgtN8);

    dim3 gGrid(MROWS / 128, EMBED / 128);    // 32 x 8
    mha_gemm_bf16_wmma<<<gGrid, 256, 0, stream>>>(Xq, Wqb, bq, (void*)Qh, 1);
    mha_gemm_bf16_wmma<<<gGrid, 256, 0, stream>>>(Xk, Wkb, bk, (void*)Kh, 1);
    mha_gemm_bf16_wmma<<<gGrid, 256, 0, stream>>>(Xv, Wvb, bv, (void*)Vt, 2);

    dim3 aGrid(2 * NUM_HEADS, SEQ / 128);    // 32 x 16
    mha_attention_wmma<<<aGrid, 256, 0, stream>>>(Qh, Kh, Vt, AO);

    mha_gemm_bf16_wmma<<<gGrid, 256, 0, stream>>>(AO, Wob, bo, d_out, 0);
}
